// INR_Base_18116172055162
// MI455X (gfx1250) — matrix operand order repaired and checked by reading, not compiled, not run on hardware
//
#include <hip/hip_runtime.h>

typedef __attribute__((ext_vector_type(16))) _Float16 v16h;
typedef __attribute__((ext_vector_type(8)))  float    v8f;

#define HASH_P1 2654435761u
#define HASH_P2 805459861u
#define HSIZE_HASH 524288u
#define NPTS 262144
#define BLOCK 256

union Frag16 { v16h h; uint4 q[2]; };

__global__ __launch_bounds__(BLOCK) void inr_fused_kernel(
    const float*  __restrict__ coords,   // [NPTS,3]
    const float4* __restrict__ table,    // [rows,4] but only first 2^19 rows addressed
    const float*  __restrict__ w_first,  // [64,64]
    const float*  __restrict__ w_h1,     // [64,64]
    const float*  __restrict__ w_h2,     // [64,64]
    const float*  __restrict__ w_last,   // [16,64]
    float*        __restrict__ out)      // [NPTS,1]
{
    __shared__ _Float16 feat[BLOCK * 64];      // 32 KB activation tile
    __shared__ _Float16 wh[3][64 * 64];        // 24 KB hidden weights (fp16)
    __shared__ _Float16 wl[16 * 64];           //  2 KB last-layer weights

    const int tid = threadIdx.x;

    // ---- stage weights into LDS as fp16 ----
    for (int i = tid; i < 64 * 64; i += BLOCK) {
        wh[0][i] = (_Float16)w_first[i];
        wh[1][i] = (_Float16)w_h1[i];
        wh[2][i] = (_Float16)w_h2[i];
    }
    for (int i = tid; i < 16 * 64; i += BLOCK) wl[i] = (_Float16)w_last[i];

    // ---- hash-grid encode: one point per thread ----
    const int gp = blockIdx.x * BLOCK + tid;
    const float cx = coords[3 * gp + 0];
    const float cy = coords[3 * gp + 1];
    const float cz = coords[3 * gp + 2];
    _Float16* frow = &feat[tid * 64];

    // dense levels 0..2 (res = 16,32,64; hsize == res^3, power of two)
    #pragma unroll
    for (int l = 0; l < 3; ++l) {
        const uint32_t res  = 16u << l;
        const uint32_t mask = res * res * res - 1u;
        const float scale = (float)(res - 1u);
        const float px = cx * scale + 0.5f;
        const float py = cy * scale + 0.5f;
        const float pz = cz * scale + 0.5f;
        const float flx = floorf(px), fly = floorf(py), flz = floorf(pz);
        const uint32_t ix = (uint32_t)(int)flx, iy = (uint32_t)(int)fly, iz = (uint32_t)(int)flz;
        const float fx = px - flx, fy = py - fly, fz = pz - flz;
        float a0 = 0.f, a1 = 0.f, a2 = 0.f, a3 = 0.f;
        #pragma unroll
        for (int c = 0; c < 8; ++c) {
            const uint32_t ox = (c >> 2) & 1u, oy = (c >> 1) & 1u, oz = c & 1u;
            const uint32_t h = (ix + ox) + (iy + oy) * res + (iz + oz) * res * res;
            const uint32_t cell = h & mask;
            const float w = (ox ? fx : 1.f - fx) * (oy ? fy : 1.f - fy) * (oz ? fz : 1.f - fz);
            const float4 e = table[cell];
            a0 = fmaf(w, e.x, a0); a1 = fmaf(w, e.y, a1);
            a2 = fmaf(w, e.z, a2); a3 = fmaf(w, e.w, a3);
        }
        frow[l * 4 + 0] = (_Float16)a0; frow[l * 4 + 1] = (_Float16)a1;
        frow[l * 4 + 2] = (_Float16)a2; frow[l * 4 + 3] = (_Float16)a3;
    }

    // hashed levels 3..15 (hsize = 2^19)
    for (int l = 3; l < 16; ++l) {
        const float scale = (float)((16u << l) - 1u);
        const float px = cx * scale + 0.5f;
        const float py = cy * scale + 0.5f;
        const float pz = cz * scale + 0.5f;
        const float flx = floorf(px), fly = floorf(py), flz = floorf(pz);
        const uint32_t ix = (uint32_t)(int)flx, iy = (uint32_t)(int)fly, iz = (uint32_t)(int)flz;
        const float fx = px - flx, fy = py - fly, fz = pz - flz;
        float a0 = 0.f, a1 = 0.f, a2 = 0.f, a3 = 0.f;
        #pragma unroll
        for (int c = 0; c < 8; ++c) {
            const uint32_t ox = (c >> 2) & 1u, oy = (c >> 1) & 1u, oz = c & 1u;
            const uint32_t h = (ix + ox) ^ ((iy + oy) * HASH_P1) ^ ((iz + oz) * HASH_P2);
            const uint32_t cell = h & (HSIZE_HASH - 1u);
            const float w = (ox ? fx : 1.f - fx) * (oy ? fy : 1.f - fy) * (oz ? fz : 1.f - fz);
            const float4 e = table[cell];
            a0 = fmaf(w, e.x, a0); a1 = fmaf(w, e.y, a1);
            a2 = fmaf(w, e.z, a2); a3 = fmaf(w, e.w, a3);
        }
        frow[l * 4 + 0] = (_Float16)a0; frow[l * 4 + 1] = (_Float16)a1;
        frow[l * 4 + 2] = (_Float16)a2; frow[l * 4 + 3] = (_Float16)a3;
    }

    __syncthreads();

    // ---- fp16 WMMA MLP: each wave runs two 16-point tiles ----
    const int lane = tid & 31;
    const int wv   = tid >> 5;
    const int grp  = lane >> 4;   // 0: lanes 0-15, 1: lanes 16-31
    const int lm   = lane & 15;

    #pragma unroll
    for (int tt = 0; tt < 2; ++tt) {
        const int t = wv * 2 + tt;           // 16-point tile index (rows t*16 .. t*16+15)

        // three hidden layers, in-place on feat
        for (int l = 0; l < 3; ++l) {
            Frag16 A0, A1;
            const _Float16* fb = &feat[(t * 16 + lm) * 64];
            // 16-bit A layout: elem e -> K = kc*32 + (e<8 ? grp*8+e : 16+grp*8+(e-8))
            A0.q[0] = *(const uint4*)(fb + 0  + grp * 8);
            A0.q[1] = *(const uint4*)(fb + 16 + grp * 8);
            A1.q[0] = *(const uint4*)(fb + 32 + grp * 8);
            A1.q[1] = *(const uint4*)(fb + 48 + grp * 8);
            #pragma unroll
            for (int nt = 0; nt < 4; ++nt) {
                // B[k][n] = W[nt*16+lm][k]; elem e -> K = kc*32 + (e<8 ? grp*8+e : 16+grp*8+(e-8))
                Frag16 B0, B1;
                const _Float16* wb = &wh[l][(nt * 16 + lm) * 64];
                B0.q[0] = *(const uint4*)(wb + 0  + grp * 8);
                B0.q[1] = *(const uint4*)(wb + 16 + grp * 8);
                B1.q[0] = *(const uint4*)(wb + 32 + grp * 8);
                B1.q[1] = *(const uint4*)(wb + 48 + grp * 8);
                v8f acc = {};
                acc = __builtin_amdgcn_wmma_f32_16x16x32_f16(false, A0.h, false, B0.h,
                                                             (short)0, acc, false, false);
                acc = __builtin_amdgcn_wmma_f32_16x16x32_f16(false, A1.h, false, B1.h,
                                                             (short)0, acc, false, false);
                // ReLU + fp16 store back in place (A already in registers; DS in-order per wave)
                #pragma unroll
                for (int r = 0; r < 8; ++r) {
                    const float v = fmaxf(acc[r], 0.f);
                    feat[(t * 16 + grp * 8 + r) * 64 + nt * 16 + lm] = (_Float16)v;
                }
            }
        }

        // last layer: only N-tile 0 (w_last is 16x64), keep output column 0
        {
            Frag16 A0, A1;
            const _Float16* fb = &feat[(t * 16 + lm) * 64];
            A0.q[0] = *(const uint4*)(fb + 0  + grp * 8);
            A0.q[1] = *(const uint4*)(fb + 16 + grp * 8);
            A1.q[0] = *(const uint4*)(fb + 32 + grp * 8);
            A1.q[1] = *(const uint4*)(fb + 48 + grp * 8);
            Frag16 B0, B1;
            const _Float16* wb = &wl[lm * 64];
            B0.q[0] = *(const uint4*)(wb + 0  + grp * 8);
            B0.q[1] = *(const uint4*)(wb + 16 + grp * 8);
            B1.q[0] = *(const uint4*)(wb + 32 + grp * 8);
            B1.q[1] = *(const uint4*)(wb + 48 + grp * 8);
            v8f acc = {};
            acc = __builtin_amdgcn_wmma_f32_16x16x32_f16(false, A0.h, false, B0.h,
                                                         (short)0, acc, false, false);
            acc = __builtin_amdgcn_wmma_f32_16x16x32_f16(false, A1.h, false, B1.h,
                                                         (short)0, acc, false, false);
            // D layout: elem r, lane L -> M = r + (L/16)*8, N = L%16. Keep N==0.
            if (lm == 0) {
                float4 s0 = { acc[0], acc[1], acc[2], acc[3] };
                float4 s1 = { acc[4], acc[5], acc[6], acc[7] };
                float* ob = out + blockIdx.x * BLOCK + t * 16 + grp * 8;
                *(float4*)(ob + 0) = s0;
                *(float4*)(ob + 4) = s1;
            }
        }
    }
}

extern "C" void kernel_launch(void* const* d_in, const int* in_sizes, int n_in,
                              void* d_out, int out_size, void* d_ws, size_t ws_size,
                              hipStream_t stream) {
    const float*  coords  = (const float*)d_in[0];
    const float4* table   = (const float4*)d_in[1];
    const float*  w_first = (const float*)d_in[2];
    const float*  w_h1    = (const float*)d_in[3];
    const float*  w_h2    = (const float*)d_in[4];
    const float*  w_last  = (const float*)d_in[5];
    float*        out     = (float*)d_out;

    const int nblocks = NPTS / BLOCK;   // 1024
    inr_fused_kernel<<<nblocks, BLOCK, 0, stream>>>(coords, table, w_first, w_h1,
                                                    w_h2, w_last, out);
}
